// ContextEncoder_88854283420019
// MI455X (gfx1250) — matrix operand order repaired and checked by reading, not compiled, not run on hardware
//
#include <hip/hip_runtime.h>
#include <hip/hip_bf16.h>
#include <math.h>

// ---------------------------------------------------------------------------
// ContextEncoder (GVP-style GNN) for MI455X / gfx1250.
//  * all channel contractions on v_wmma_f32_16x16x32_bf16 (bf16 in, f32 acc)
//  * GEMM: 2x2 tiles per wave (A/B fragment reuse, 4 wmma per 4 frag loads)
//    + software-pipelined K loop (double fragment buffers -> loads overlap wmma)
//  * edge pipeline chunked to 32K edges so E-sized intermediates fit in the
//    192MB global L2 (per-edge working set ~4.9KB -> ~160MB/chunk)
//  * rank-1 algebra for the edge_gv vector path (unit (x) w_vexp is rank-1)
//  * gathers folded into GEMM via row-index indirection; scatter via atomics
// Workspace use: ~0.29 GB (chunked). Assumes ws_size is at least that.
// ---------------------------------------------------------------------------

typedef unsigned short ushort_t;
typedef __attribute__((ext_vector_type(16))) __bf16        v16bf;
typedef __attribute__((ext_vector_type(8)))  float         v8f;
typedef __attribute__((ext_vector_type(8)))  unsigned int  v8u;

constexpr int kN  = 10000;     // nodes
constexpr int kE  = 160000;    // edges
constexpr int kFS = 256;       // scalar feat
constexpr int kFV = 64;        // vector channels
constexpr int kEC = 64;        // edge channels
constexpr int kNG = 60;        // gaussians
constexpr int kNET = 4;        // edge types
constexpr int kNB = 6;         // blocks
constexpr int kCHUNK = 32768;  // edges per chunk (L2-resident)
constexpr float kCUT = 10.0f;
constexpr float kSLOPE = 0.2f;
constexpr float kEPS = 1e-6f;

#define ACT_NONE 0
#define ACT_LRELU 1
#define ACT_SIG 2

__device__ __forceinline__ ushort_t f2bf(float f) {
  unsigned int u = __builtin_bit_cast(unsigned int, f);
  u += 0x7FFFu + ((u >> 16) & 1u);
  return (ushort_t)(u >> 16);
}

struct Frag { uint4 lo, hi; };
__device__ __forceinline__ Frag ldA(const ushort_t* p) {   // ISA 16-bit A layout
  Frag f; f.lo = *(const uint4*)p; f.hi = *(const uint4*)(p + 16); return f;
}
__device__ __forceinline__ Frag ldB(const ushort_t* p) {   // ISA 16-bit B layout
  Frag f; f.lo = *(const uint4*)p; f.hi = *(const uint4*)(p + 16); return f;
}
__device__ __forceinline__ v16bf f2v(const Frag& f) {
  v8u u = {f.lo.x, f.lo.y, f.lo.z, f.lo.w, f.hi.x, f.hi.y, f.hi.z, f.hi.w};
  return __builtin_bit_cast(v16bf, u);
}

// ---------------------------------------------------------------------------
// WMMA GEMM:  C[M,N] (+)= act( A[M,K] @ W[N,K]^T + bias )
// A: bf16 row-major (lda elems), optionally row-gathered through rowIdx.
// W: bf16 row-major (ldw elems).  One wave -> 32x32 output (2x2 WMMA tiles);
// 8 waves per block cover 256 rows.  K must be a multiple of 32 (true here).
// ---------------------------------------------------------------------------
__global__ __launch_bounds__(256)
void k_wmma_gemm(const ushort_t* __restrict__ A, const int* __restrict__ rowIdx,
                 const ushort_t* __restrict__ W, const float* __restrict__ bias,
                 float* __restrict__ C, int M, int Nn, int K,
                 int lda, int ldw, int ldc, int accum, int act)
{
  const int lane = threadIdx.x & 31;
  const int wave = threadIdx.x >> 5;
  const int mT0 = (blockIdx.x * 8 + wave) * 32;
  const int mT1 = mT0 + 16;
  const int nT0 = blockIdx.y * 32;
  const int nT1 = nT0 + 16;
  if (mT0 >= M) return;                 // uniform per wave: EXEC stays full

  const int half = lane >> 4;           // 0 | 1
  const int l15  = lane & 15;

  auto arowp = [&](int mT) -> const ushort_t* {
    int mRow = mT + l15;
    int aRow = mRow < M ? mRow : M - 1;
    if (rowIdx) aRow = rowIdx[aRow];
    return A + (size_t)aRow * lda + half * 8;
  };
  auto wrowp = [&](int nT) -> const ushort_t* {
    int nCol = nT + l15;
    int wRow = nCol < Nn ? nCol : Nn - 1;
    return W + (size_t)wRow * ldw + half * 8;
  };
  const ushort_t* a0p = arowp(mT0);
  const ushort_t* a1p = arowp(mT1);
  const ushort_t* b0p = wrowp(nT0);
  const ushort_t* b1p = wrowp(nT1);

  const v8f vz = {0.f, 0.f, 0.f, 0.f, 0.f, 0.f, 0.f, 0.f};
  v8f acc00 = vz, acc01 = vz, acc10 = vz, acc11 = vz;
  auto loadC = [&](int mT, int nT) -> v8f {
    v8f a = vz;
    int cCol = nT + l15;
    if (cCol < Nn) {
      int cRow = mT + half * 8;
      for (int r = 0; r < 8; ++r) {
        int rr = cRow + r;
        if (rr < M) a[r] = C[(size_t)rr * ldc + cCol];
      }
    }
    return a;
  };
  if (accum) {
    acc00 = loadC(mT0, nT0); acc01 = loadC(mT0, nT1);
    acc10 = loadC(mT1, nT0); acc11 = loadC(mT1, nT1);
  }

  // software-pipelined K loop: next k-step loads overlap current 4 WMMAs
  Frag a0 = ldA(a0p), a1 = ldA(a1p), b0 = ldB(b0p), b1 = ldB(b1p);
  for (int k0 = 0; k0 < K; k0 += 32) {
    const int kn = k0 + 32;
    const bool more = kn < K;
    Frag na0{}, na1{}, nb0{}, nb1{};
    if (more) {
      if (kn + 64 < K) {                       // -> global_prefetch_b8
        __builtin_prefetch((const void*)(a0p + kn + 64), 0, 1);
        __builtin_prefetch((const void*)(a1p + kn + 64), 0, 1);
      }
      na0 = ldA(a0p + kn); na1 = ldA(a1p + kn);
      nb0 = ldB(b0p + kn); nb1 = ldB(b1p + kn);
    }
    v16bf va0 = f2v(a0), va1 = f2v(a1), vb0 = f2v(b0), vb1 = f2v(b1);
    acc00 = __builtin_amdgcn_wmma_f32_16x16x32_bf16(false, va0, false, vb0,
                                                    (short)0, acc00, false, false);
    acc01 = __builtin_amdgcn_wmma_f32_16x16x32_bf16(false, va0, false, vb1,
                                                    (short)0, acc01, false, false);
    acc10 = __builtin_amdgcn_wmma_f32_16x16x32_bf16(false, va1, false, vb0,
                                                    (short)0, acc10, false, false);
    acc11 = __builtin_amdgcn_wmma_f32_16x16x32_bf16(false, va1, false, vb1,
                                                    (short)0, acc11, false, false);
    if (more) { a0 = na0; a1 = na1; b0 = nb0; b1 = nb1; }
  }

  auto storeC = [&](v8f a, int mT, int nT) {
    int cCol = nT + l15;
    if (cCol >= Nn) return;
    float bb = bias ? bias[cCol] : 0.0f;
    int cRow = mT + half * 8;
    for (int r = 0; r < 8; ++r) {
      int rr = cRow + r;
      if (rr >= M) continue;
      float v = a[r] + bb;
      if (act == ACT_LRELU)    v = v >= 0.f ? v : 0.01f * v;
      else if (act == ACT_SIG) v = 1.0f / (1.0f + __expf(-v));
      C[(size_t)rr * ldc + cCol] = v;
    }
  };
  storeC(acc00, mT0, nT0); storeC(acc01, mT0, nT1);
  storeC(acc10, mT1, nT0); storeC(acc11, mT1, nT1);
}

// --------------------------- elementwise kernels ---------------------------
__global__ __launch_bounds__(256) void k_copy(const float* s, float* d, int n) {
  int i = blockIdx.x * 256 + threadIdx.x; if (i < n) d[i] = s[i];
}
__global__ __launch_bounds__(256) void k_zero(float* d, int n) {
  int i = blockIdx.x * 256 + threadIdx.x; if (i < n) d[i] = 0.f;
}
__global__ __launch_bounds__(256) void k_f2bf(const float* s, ushort_t* d, int n) {
  int i = blockIdx.x * 256 + threadIdx.x; if (i < n) d[i] = f2bf(s[i]);
}
__global__ __launch_bounds__(256) void k_split3(const float* v, ushort_t* p0,
                                                ushort_t* p1, ushort_t* p2, int n) {
  int i = blockIdx.x * 256 + threadIdx.x; if (i >= n) return;
  p0[i] = f2bf(v[i * 3 + 0]); p1[i] = f2bf(v[i * 3 + 1]); p2[i] = f2bf(v[i * 3 + 2]);
}
__global__ __launch_bounds__(256) void k_lrelu_bf(const float* s, ushort_t* d, int n) {
  int i = blockIdx.x * 256 + threadIdx.x; if (i >= n) return;
  float v = s[i]; d[i] = f2bf(v >= 0.f ? v : 0.01f * v);
}
__global__ __launch_bounds__(256) void k_vnorm_planes(
    const float* p0, const float* p1, const float* p2,
    ushort_t* vn, ushort_t* b0, ushort_t* b1, ushort_t* b2, int n) {
  int i = blockIdx.x * 256 + threadIdx.x; if (i >= n) return;
  float a = p0[i], b = p1[i], c = p2[i];
  vn[i] = f2bf(sqrtf(a * a + b * b + c * c));
  b0[i] = f2bf(a); b1[i] = f2bf(b); b2[i] = f2bf(c);
}
__global__ __launch_bounds__(256) void k_gatemul(const float* g, float* p0,
                                                 float* p1, float* p2, int n) {
  int i = blockIdx.x * 256 + threadIdx.x; if (i >= n) return;
  float gv = g[i]; p0[i] *= gv; p1[i] *= gv; p2[i] *= gv;
}
__global__ __launch_bounds__(256) void k_res_sca(float* h, const float* d, int n) {
  int i = blockIdx.x * 256 + threadIdx.x; if (i < n) h[i] += d[i];
}
__global__ __launch_bounds__(256) void k_res_vec(float* h, const float* g,
    const float* w0, const float* w1, const float* w2, int n) {
  int i = blockIdx.x * 256 + threadIdx.x; if (i >= n) return;
  float gv = g[i];
  h[i * 3 + 0] += gv * w0[i]; h[i * 3 + 1] += gv * w1[i]; h[i * 3 + 2] += gv * w2[i];
}

// --------------------------- edge geometry ---------------------------------
__global__ __launch_bounds__(256) void k_edge_geom(
    const float* __restrict__ pos, const float* __restrict__ efeat,
    const int* __restrict__ eidx, float* dist, float* u2, float* ccut,
    float* unitv, ushort_t* esfB)
{
  int i = blockIdx.x * 256 + threadIdx.x;
  if (i >= kE * kEC) return;
  int e = i >> 6, c = i & 63;
  int r = eidx[e], cl = eidx[kE + e];
  float dx = pos[r * 3 + 0] - pos[cl * 3 + 0];
  float dy = pos[r * 3 + 1] - pos[cl * 3 + 1];
  float dz = pos[r * 3 + 2] - pos[cl * 3 + 2];
  float d = sqrtf(dx * dx + dy * dy + dz * dz);
  float inv = 1.0f / (d + 1e-7f);
  float fea;
  if (c < kNG) {
    const float step = kCUT / (float)(kNG - 1);
    float t = d - step * (float)c;
    const float coeff = -0.5f / (step * step);
    fea = __expf(coeff * t * t);
  } else {
    fea = efeat[e * kNET + (c - kNG)];
  }
  esfB[i] = f2bf(fea);
  if (c == 0) dist[e] = d;
  else if (c == 1) {
    float cc = 0.5f * (__cosf(d * 3.14159265358979f / kCUT) + 1.0f);
    ccut[e] = (d <= kCUT) ? cc : 0.0f;
  } else if (c == 2) {
    float ux = dx * inv, uy = dy * inv, uz = dz * inv;
    u2[e] = ux * ux + uy * uy + uz * uz;
  } else if (c < 6) {
    unitv[e * 3 + (c - 3)] = ((c == 3) ? dx : (c == 4) ? dy : dz) * inv;
  }
}

// q = Wv1*w ; r = Wv2*q ; t = Ws[:, :64]*|q|   (rank-1 edge_gv precompute)
__global__ void k_matvec_small(const float* Wv1, const float* Wv2,
                               const float* Ws, const float* w, float* qrt)
{
  __shared__ float q[64];
  int o = threadIdx.x;
  float qv = 0.f;
  for (int c = 0; c < 64; ++c) qv += Wv1[o * 64 + c] * w[c];
  q[o] = qv; __syncthreads();
  float rv = 0.f, tv = 0.f;
  for (int c = 0; c < 64; ++c) {
    rv += Wv2[o * 64 + c] * q[c];
    tv += Ws[o * 128 + c] * fabsf(q[c]);
  }
  qrt[o] = qv; qrt[64 + o] = rv; qrt[128 + o] = tv;
}

__global__ __launch_bounds__(256) void k_presca(const float* u2, const float* t,
                                                float* out, int n) {
  int i = blockIdx.x * 256 + threadIdx.x; if (i >= n) return;
  out[i] = sqrtf(u2[i >> 6]) * t[i & 63];
}
__global__ __launch_bounds__(256) void k_evcoef(const float* g, const float* r,
                                                ushort_t* out, int n) {
  int i = blockIdx.x * 256 + threadIdx.x; if (i >= n) return;
  out[i] = f2bf(g[i] * r[i & 63]);
}
__global__ __launch_bounds__(256) void k_fcoef(const float* g, const float* s,
    const float* r, const float* u2, ushort_t* out, int n) {
  int i = blockIdx.x * 256 + threadIdx.x; if (i >= n) return;
  float c = g[i] * r[i & 63];
  float sv = s[i], uu = u2[i >> 6];
  float dot = c * sv * uu, dsq = sv * sv * uu;
  float neg = c - (dot / (dsq + kEPS)) * sv;
  float res = (dot >= 0.f) ? c : neg;
  out[i] = f2bf(kSLOPE * c + (1.f - kSLOPE) * res);
}
__global__ __launch_bounds__(256) void k_ysca(const float* nsca, const int* col,
    const float* sl, ushort_t* out, int n) {
  int i = blockIdx.x * 256 + threadIdx.x; if (i >= n) return;
  int e = i >> 8, k = i & 255;
  out[i] = f2bf(nsca[(size_t)col[e] * kFS + k] * sl[i]);
}
__global__ __launch_bounds__(256) void k_yvec(const float* e2n, const float* n2e,
    const float* g, const float* unitv, const int* col,
    const float* nv0, const float* nv1, const float* nv2,
    ushort_t* y0, ushort_t* y1, ushort_t* y2, int n) {
  int i = blockIdx.x * 256 + threadIdx.x; if (i >= n) return;
  int e = i >> 6, o = i & 63;
  int cl = col[e];
  float a = e2n[i];
  float b = n2e[i] * g[i];
  float u0 = unitv[e * 3 + 0], u1 = unitv[e * 3 + 1], u2v = unitv[e * 3 + 2];
  y0[i] = f2bf(a * nv0[cl * 64 + o] + b * u0);
  y1[i] = f2bf(a * nv1[cl * 64 + o] + b * u1);
  y2[i] = f2bf(a * nv2[cl * 64 + o] + b * u2v);
}
__global__ __launch_bounds__(256) void k_scatter_sca(const float* msca,
    const float* ccut, const int* row, float* agg, int n) {
  int i = blockIdx.x * 256 + threadIdx.x; if (i >= n) return;
  int e = i >> 8, k = i & 255;
  atomicAdd(&agg[(size_t)row[e] * kFS + k], ccut[e] * msca[i]);
}
__global__ __launch_bounds__(256) void k_scatter_vec(const float* gate,
    const float* m0, const float* m1, const float* m2,
    const float* ccut, const int* row, float* agg, int n) {
  int i = blockIdx.x * 256 + threadIdx.x; if (i >= n) return;
  int e = i >> 6, o = i & 63;
  float cc = ccut[e] * gate[i];
  size_t base = (size_t)row[e] * 192 + o * 3;
  atomicAdd(&agg[base + 0], cc * m0[i]);
  atomicAdd(&agg[base + 1], cc * m1[i]);
  atomicAdd(&agg[base + 2], cc * m2[i]);
}

// --------------------------- layer norms -----------------------------------
__global__ __launch_bounds__(256) void k_ln_sca(const float* c, const float* agg,
    const float* g, const float* b, ushort_t* outBf) {
  __shared__ float s1[256], s2[256];
  int n = blockIdx.x, t = threadIdx.x;
  float x = c[(size_t)n * kFS + t] + agg[(size_t)n * kFS + t];
  s1[t] = x; s2[t] = x * x; __syncthreads();
  for (int st = 128; st > 0; st >>= 1) {
    if (t < st) { s1[t] += s1[t + st]; s2[t] += s2[t + st]; } __syncthreads();
  }
  float mean = s1[0] * (1.0f / kFS);
  float var = s2[0] * (1.0f / kFS) - mean * mean;
  float y = (x - mean) * rsqrtf(var + 1e-5f) * g[t] + b[t];
  y = y >= 0.f ? y : 0.01f * y;   // leaky after LN
  outBf[(size_t)n * kFS + t] = f2bf(y);
}
__global__ __launch_bounds__(256) void k_ln_vec(const float* gateC,
    const float* c0, const float* c1, const float* c2, const float* agg,
    const float* g, const float* b,
    float* f0, float* f1, float* f2p, ushort_t* b0, ushort_t* b1, ushort_t* b2) {
  __shared__ float s1[256], s2[256];
  int n = blockIdx.x, t = threadIdx.x;
  int o = t / 3, d = t % 3;
  float x = 0.f;
  if (t < 192) {
    const float* cp = (d == 0) ? c0 : (d == 1) ? c1 : c2;
    x = gateC[n * 64 + o] * cp[n * 64 + o] + agg[(size_t)n * 192 + t];
  }
  s1[t] = x; s2[t] = x * x; __syncthreads();
  for (int st = 128; st > 0; st >>= 1) {
    if (t < st) { s1[t] += s1[t + st]; s2[t] += s2[t + st]; } __syncthreads();
  }
  float mean = s1[0] * (1.0f / 192.f);
  float var = s2[0] * (1.0f / 192.f) - mean * mean;
  if (t < 192) {
    float y = (x - mean) * rsqrtf(var + 1e-5f) * g[t] + b[t];
    int idx = n * 64 + o;
    if (d == 0) { f0[idx] = y; b0[idx] = f2bf(y); }
    else if (d == 1) { f1[idx] = y; b1[idx] = f2bf(y); }
    else { f2p[idx] = y; b2[idx] = f2bf(y); }
  }
}
__global__ __launch_bounds__(256) void k_vnlrelu(
    const float* x0, const float* x1, const float* x2,
    const float* d0, const float* d1, const float* d2,
    ushort_t* o0, ushort_t* o1, ushort_t* o2, int n) {
  int i = blockIdx.x * 256 + threadIdx.x; if (i >= n) return;
  float a0 = x0[i], a1 = x1[i], a2 = x2[i];
  float e0 = d0[i], e1 = d1[i], e2 = d2[i];
  float dot = a0 * e0 + a1 * e1 + a2 * e2;
  float dsq = e0 * e0 + e1 * e1 + e2 * e2;
  float t = dot / (dsq + kEPS);
  float r0 = dot >= 0.f ? a0 : a0 - t * e0;
  float r1 = dot >= 0.f ? a1 : a1 - t * e1;
  float r2 = dot >= 0.f ? a2 : a2 - t * e2;
  o0[i] = f2bf(kSLOPE * a0 + (1.f - kSLOPE) * r0);
  o1[i] = f2bf(kSLOPE * a1 + (1.f - kSLOPE) * r1);
  o2[i] = f2bf(kSLOPE * a2 + (1.f - kSLOPE) * r2);
}

// ---------------------------------------------------------------------------
extern "C" void kernel_launch(void* const* d_in, const int* in_sizes, int n_in,
                              void* d_out, int out_size, void* d_ws, size_t ws_size,
                              hipStream_t stream) {
  if (n_in < 5 + kNB * 39) return;
  const float* in_sca  = (const float*)d_in[0];
  const float* in_vec  = (const float*)d_in[1];
  const float* pos     = (const float*)d_in[2];
  const float* efeat   = (const float*)d_in[3];
  const int*   eidx    = (const int*)d_in[4];
  const float* P[kNB][39];
  for (int b = 0; b < kNB; ++b)
    for (int j = 0; j < 39; ++j) P[b][j] = (const float*)d_in[5 + b * 39 + j];
  const int* erow = eidx;          // edge_index[0]
  const int* ecol = eidx + kE;     // edge_index[1]

  float* h_sca = (float*)d_out;
  float* h_vec = (float*)d_out + (size_t)kN * kFS;

  char* wsb = (char*)d_ws;
  size_t wsOff = 0;
  auto balloc = [&](size_t bytes) -> char* {
    size_t a = (wsOff + 255) & ~(size_t)255;
    wsOff = a + bytes;
    return wsb + a;
  };
  auto flat = [](int n) { return dim3((unsigned)((n + 255) / 256)); };
  auto conv = [&](const float* src, int n) -> const ushort_t* {
    ushort_t* dst = (ushort_t*)balloc((size_t)n * 2);
    k_f2bf<<<flat(n), 256, 0, stream>>>(src, dst, n);
    return dst;
  };
  auto gemm = [&](const ushort_t* A, const int* rows, const ushort_t* W,
                  const float* bias, float* C, int M, int Nn, int K,
                  int lda, int ldw, int ldc, int accum, int act) {
    dim3 g((unsigned)((M + 255) / 256), (unsigned)((Nn + 31) / 32));
    k_wmma_gemm<<<g, 256, 0, stream>>>(A, rows, W, bias, C, M, Nn, K,
                                       lda, ldw, ldc, accum, act);
  };

  // ---- weights -> bf16 (deterministic, every call) ----
  struct BW {
    const ushort_t *ngWv1, *ngWv2, *ngWs, *ngWg;
    const ushort_t *egWs, *egWg, *eav, *scaLinW, *e2nW, *n2eW, *evn;
    const ushort_t *mgWv1, *mgWv2, *mgWs, *mgWg;
    const ushort_t *ctWv1, *ctWv2, *ctWs, *ctWg;
    const ushort_t *actv;
    const ushort_t *otWv1, *otWv2, *otWs, *otWg;
  } BWs[kNB];
  for (int b = 0; b < kNB; ++b) {
    BWs[b].ngWv1 = conv(P[b][1], 4096);   BWs[b].ngWv2 = conv(P[b][2], 4096);
    BWs[b].ngWs  = conv(P[b][3], 81920);  BWs[b].ngWg  = conv(P[b][4], 16384);
    BWs[b].egWs  = conv(P[b][8], 8192);   BWs[b].egWg  = conv(P[b][9], 4096);
    BWs[b].eav   = conv(P[b][11], 4096);  BWs[b].scaLinW = conv(P[b][12], 16384);
    BWs[b].e2nW  = conv(P[b][14], 4096);  BWs[b].n2eW  = conv(P[b][16], 16384);
    BWs[b].evn   = conv(P[b][18], 4096);
    BWs[b].mgWv1 = conv(P[b][19], 4096);  BWs[b].mgWv2 = conv(P[b][20], 4096);
    BWs[b].mgWs  = conv(P[b][21], 81920); BWs[b].mgWg  = conv(P[b][22], 16384);
    BWs[b].ctWv1 = conv(P[b][24], 4096);  BWs[b].ctWv2 = conv(P[b][25], 4096);
    BWs[b].ctWs  = conv(P[b][26], 81920); BWs[b].ctWg  = conv(P[b][27], 16384);
    BWs[b].actv  = conv(P[b][29], 4096);
    BWs[b].otWv1 = conv(P[b][30], 4096);  BWs[b].otWv2 = conv(P[b][31], 4096);
    BWs[b].otWs  = conv(P[b][32], 81920); BWs[b].otWg  = conv(P[b][33], 16384);
  }

  // ---- workspace buffers ----
  float* dist  = (float*)balloc((size_t)kE * 4);
  float* u2    = (float*)balloc((size_t)kE * 4);
  float* ccut  = (float*)balloc((size_t)kE * 4);
  float* unitv = (float*)balloc((size_t)kE * 12);
  ushort_t* esfB = (ushort_t*)balloc((size_t)kE * kEC * 2);
  float* qrt = (float*)balloc(192 * 4);

  ushort_t* hscaBf = (ushort_t*)balloc((size_t)kN * kFS * 2);
  ushort_t* hvB[3]; float* vmF[3]; ushort_t* vmB[3];
  float* nA_w2[3]; float* nB_w2[3]; float* lnF[3]; ushort_t* lnB[3];
  for (int d = 0; d < 3; ++d) {
    hvB[d] = (ushort_t*)balloc((size_t)kN * kFV * 2);
    vmF[d] = (float*)balloc((size_t)kN * kFV * 4);
    vmB[d] = (ushort_t*)balloc((size_t)kN * kFV * 2);
    nA_w2[d] = (float*)balloc((size_t)kN * kFV * 4);
    nB_w2[d] = (float*)balloc((size_t)kN * kFV * 4);
    lnF[d] = (float*)balloc((size_t)kN * kFV * 4);
    lnB[d] = (ushort_t*)balloc((size_t)kN * kFV * 2);
  }
  ushort_t* vnB = (ushort_t*)balloc((size_t)kN * kFV * 2);
  float* nA_sca = (float*)balloc((size_t)kN * kFS * 4);
  ushort_t* nA_scaBf = (ushort_t*)balloc((size_t)kN * kFS * 2);
  float* nA_gate = (float*)balloc((size_t)kN * kFV * 4);
  float* nB_sca = (float*)balloc((size_t)kN * kFS * 4);
  ushort_t* nB_scaBf = (ushort_t*)balloc((size_t)kN * kFS * 2);
  float* nB_gate = (float*)balloc((size_t)kN * kFV * 4);
  ushort_t* lnscaBf = (ushort_t*)balloc((size_t)kN * kFS * 2);
  float* aggS = (float*)balloc((size_t)kN * kFS * 4);
  float* aggV = (float*)balloc((size_t)kN * 192 * 4);

  const int L0 = kCHUNK;
  float* esca  = (float*)balloc((size_t)L0 * kEC * 4);
  ushort_t* t0 = (ushort_t*)balloc((size_t)L0 * kEC * 2);
  ushort_t* t1 = (ushort_t*)balloc((size_t)L0 * kEC * 2);
  ushort_t* t2 = (ushort_t*)balloc((size_t)L0 * kEC * 2);
  float* gateE = (float*)balloc((size_t)L0 * kEC * 4);   // also gateM
  float* sE    = (float*)balloc((size_t)L0 * kEC * 4);   // also e2n out
  float* gE    = (float*)balloc((size_t)L0 * kEC * 4);
  float* n2eE  = (float*)balloc((size_t)L0 * kEC * 4);
  float* msca  = (float*)balloc((size_t)L0 * kFS * 4);   // sl then m_sca
  ushort_t* yscaBf = (ushort_t*)balloc((size_t)L0 * kFS * 2);  // y_sca then m_sca bf
  ushort_t* yvB[3]; float* ymF[3]; ushort_t* ymB[3];
  for (int d = 0; d < 3; ++d) {
    yvB[d] = (ushort_t*)balloc((size_t)L0 * kEC * 2);
    ymF[d] = (float*)balloc((size_t)L0 * kEC * 4);       // ymid, then m_vec
    ymB[d] = (ushort_t*)balloc((size_t)L0 * kEC * 2);
  }
  (void)ws_size; (void)in_sizes; (void)out_size;

  // ---- init h = inputs; geometry ----
  k_copy<<<flat(kN * kFS), 256, 0, stream>>>(in_sca, h_sca, kN * kFS);
  k_copy<<<flat(kN * 192), 256, 0, stream>>>(in_vec, h_vec, kN * 192);
  k_edge_geom<<<flat(kE * kEC), 256, 0, stream>>>(pos, efeat, eidx, dist, u2,
                                                  ccut, unitv, esfB);

  // gvlinear "front": vmid planes, vnorm, out_sca, gate, Wv2 planes
  auto gv_front = [&](int M, const ushort_t* scaBf, ushort_t* const vecB[3],
                      const ushort_t* Wv1, const ushort_t* Wv2,
                      const ushort_t* Ws, const ushort_t* Wg, const float* bg,
                      float* const vF[3], ushort_t* const vB[3], ushort_t* vn,
                      float* outSca, ushort_t* outScaBf, float* gate,
                      float* const w2F[3]) {
    for (int d = 0; d < 3; ++d)
      gemm(vecB[d], nullptr, Wv1, nullptr, vF[d], M, 64, 64, 64, 64, 64, 0, 0);
    k_vnorm_planes<<<flat(M * 64), 256, 0, stream>>>(vF[0], vF[1], vF[2], vn,
                                                     vB[0], vB[1], vB[2], M * 64);
    gemm(vn, nullptr, Ws, nullptr, outSca, M, 256, 64, 64, 320, 256, 0, 0);
    gemm(scaBf, nullptr, Ws + 64, nullptr, outSca, M, 256, 256, 256, 320, 256, 1, 0);
    k_f2bf<<<flat(M * 256), 256, 0, stream>>>(outSca, outScaBf, M * 256);
    gemm(outScaBf, nullptr, Wg, bg, gate, M, 64, 256, 256, 256, 64, 0, ACT_SIG);
    for (int d = 0; d < 3; ++d)
      gemm(vB[d], nullptr, Wv2, nullptr, w2F[d], M, 64, 64, 64, 64, 64, 0, 0);
  };

  for (int b = 0; b < kNB; ++b) {
    const BW& w = BWs[b];
    // node prep
    k_zero<<<flat(kN * kFS), 256, 0, stream>>>(aggS, kN * kFS);
    k_zero<<<flat(kN * 192), 256, 0, stream>>>(aggV, kN * 192);
    k_f2bf<<<flat(kN * kFS), 256, 0, stream>>>(h_sca, hscaBf, kN * kFS);
    k_split3<<<flat(kN * kFV), 256, 0, stream>>>(h_vec, hvB[0], hvB[1], hvB[2], kN * kFV);

    // node_gv
    gv_front(kN, hscaBf, hvB, w.ngWv1, w.ngWv2, w.ngWs, w.ngWg, P[b][5],
             vmF, vmB, vnB, nA_sca, nA_scaBf, nA_gate, nA_w2);
    k_gatemul<<<flat(kN * kFV), 256, 0, stream>>>(nA_gate, nA_w2[0], nA_w2[1],
                                                  nA_w2[2], kN * kFV);
    // rank-1 edge_gv precompute: q, r=Wv2 q, t=Ws[:, :64]|q|
    k_matvec_small<<<1, 64, 0, stream>>>(P[b][6], P[b][7], P[b][8], P[b][0], qrt);

    // ---- edge pipeline, L2-resident chunks ----
    for (int cs = 0; cs < kE; cs += kCHUNK) {
      int L = (kE - cs < kCHUNK) ? (kE - cs) : kCHUNK;
      int L64 = L * 64, L256 = L * 256;
      k_presca<<<flat(L64), 256, 0, stream>>>(u2 + cs, qrt + 128, esca, L64);
      gemm(esfB + (size_t)cs * 64, nullptr, w.egWs + 64, nullptr, esca,
           L, 64, 64, 64, 128, 64, 1, 0);
      k_f2bf<<<flat(L64), 256, 0, stream>>>(esca, t0, L64);
      gemm(t0, nullptr, w.egWg, P[b][10], gateE, L, 64, 64, 64, 64, 64, 0, ACT_SIG);
      k_evcoef<<<flat(L64), 256, 0, stream>>>(gateE, qrt + 64, t1, L64);
      gemm(t1, nullptr, w.eav, nullptr, sE, L, 64, 64, 64, 64, 64, 0, 0);
      k_fcoef<<<flat(L64), 256, 0, stream>>>(gateE, sE, qrt + 64, u2 + cs, t2, L64);
      gemm(t2, nullptr, w.evn, nullptr, gE, L, 64, 64, 64, 64, 64, 0, 0);
      k_lrelu_bf<<<flat(L64), 256, 0, stream>>>(esca, t1, L64);
      gemm(t1, nullptr, w.scaLinW, P[b][13], msca, L, 256, 64, 64, 64, 256, 0, 0);
      gemm(t1, nullptr, w.e2nW, P[b][15], sE, L, 64, 64, 64, 64, 64, 0, 0);
      gemm(nA_scaBf, ecol + cs, w.n2eW, P[b][17], n2eE, L, 64, 256, 256, 256, 64, 0, 0);
      k_ysca<<<flat(L256), 256, 0, stream>>>(nA_sca, ecol + cs, msca, yscaBf, L256);
      k_yvec<<<flat(L64), 256, 0, stream>>>(sE, n2eE, gE, unitv + (size_t)cs * 3,
          ecol + cs, nA_w2[0], nA_w2[1], nA_w2[2], yvB[0], yvB[1], yvB[2], L64);
      // msg_out_gv
      for (int d = 0; d < 3; ++d)
        gemm(yvB[d], nullptr, w.mgWv1, nullptr, ymF[d], L, 64, 64, 64, 64, 64, 0, 0);
      k_vnorm_planes<<<flat(L64), 256, 0, stream>>>(ymF[0], ymF[1], ymF[2], t0,
                                                    ymB[0], ymB[1], ymB[2], L64);
      gemm(t0, nullptr, w.mgWs, nullptr, msca, L, 256, 64, 64, 320, 256, 0, 0);
      gemm(yscaBf, nullptr, w.mgWs + 64, nullptr, msca, L, 256, 256, 256, 320, 256, 1, 0);
      k_f2bf<<<flat(L256), 256, 0, stream>>>(msca, yscaBf, L256);
      gemm(yscaBf, nullptr, w.mgWg, P[b][23], gateE, L, 64, 256, 256, 256, 64, 0, ACT_SIG);
      for (int d = 0; d < 3; ++d)
        gemm(ymB[d], nullptr, w.mgWv2, nullptr, ymF[d], L, 64, 64, 64, 64, 64, 0, 0);
      // cutoff-scaled scatter-sum
      k_scatter_sca<<<flat(L256), 256, 0, stream>>>(msca, ccut + cs, erow + cs,
                                                    aggS, L256);
      k_scatter_vec<<<flat(L64), 256, 0, stream>>>(gateE, ymF[0], ymF[1], ymF[2],
                                                   ccut + cs, erow + cs, aggV, L64);
    }

    // centroid + aggregation + LNs
    gv_front(kN, hscaBf, hvB, w.ctWv1, w.ctWv2, w.ctWs, w.ctWg, P[b][28],
             vmF, vmB, vnB, nB_sca, nB_scaBf, nB_gate, nB_w2);
    k_ln_sca<<<kN, 256, 0, stream>>>(nB_sca, aggS, P[b][35], P[b][36], lnscaBf);
    k_ln_vec<<<kN, 256, 0, stream>>>(nB_gate, nB_w2[0], nB_w2[1], nB_w2[2], aggV,
        P[b][37], P[b][38], lnF[0], lnF[1], lnF[2], lnB[0], lnB[1], lnB[2]);
    // vn_leaky_relu(act_vec)
    for (int d = 0; d < 3; ++d)
      gemm(lnB[d], nullptr, w.actv, nullptr, vmF[d], kN, 64, 64, 64, 64, 64, 0, 0);
    k_vnlrelu<<<flat(kN * kFV), 256, 0, stream>>>(lnF[0], lnF[1], lnF[2],
        vmF[0], vmF[1], vmF[2], lnB[0], lnB[1], lnB[2], kN * kFV);
    // out_tf + residual
    gv_front(kN, lnscaBf, lnB, w.otWv1, w.otWv2, w.otWs, w.otWg, P[b][34],
             vmF, vmB, vnB, nB_sca, nB_scaBf, nB_gate, nB_w2);
    k_res_sca<<<flat(kN * kFS), 256, 0, stream>>>(h_sca, nB_sca, kN * kFS);
    k_res_vec<<<flat(kN * kFV), 256, 0, stream>>>(h_vec, nB_gate, nB_w2[0],
                                                  nB_w2[1], nB_w2[2], kN * kFV);
  }
}
